// MultiHeadAttention_69449621176676
// MI455X (gfx1250) — matrix operand order repaired and checked by reading, not compiled, not run on hardware
//
#include <hip/hip_runtime.h>
#include <hip/hip_bf16.h>

// ---------------- problem constants ----------------
#define BATCH   32
#define QLEN    8
#define KLEN    2048
#define DIN     256
#define NHEAD   8
#define DH      32          // DIN / NHEAD
#define BH      (BATCH * NHEAD)   // 256

typedef __attribute__((ext_vector_type(16))) __bf16    v16bf;
typedef __attribute__((ext_vector_type(8)))  float     v8f;
typedef __attribute__((ext_vector_type(4)))  unsigned  v4u;
typedef __attribute__((ext_vector_type(8)))  int       v8i32;
typedef __attribute__((ext_vector_type(4)))  int       v4i32;

// ---------------------------------------------------------------------------
// fast device math on the transcendental pipe
// ---------------------------------------------------------------------------
__device__ __forceinline__ float fast_exp(float x) {
    return __builtin_amdgcn_exp2f(x * 1.4426950408889634f);
}
__device__ __forceinline__ float fast_tanh(float x) {
    // tanh(x) = 1 - 2 / (e^{2x} + 1); saturates correctly at +/-1 via exp2->inf/0
    float t = __builtin_amdgcn_exp2f(x * 2.8853900817779268f);
    return 1.0f - 2.0f * __builtin_amdgcn_rcpf(t + 1.0f);
}

// ---------------------------------------------------------------------------
// f32 -> bf16 conversion (coalesced)
// ---------------------------------------------------------------------------
__global__ void f32_to_bf16_kernel(const float* __restrict__ in,
                                   __bf16* __restrict__ out, int n) {
    int i = blockIdx.x * blockDim.x + threadIdx.x;
    if (i < n) out[i] = (__bf16)in[i];
}

// ---------------------------------------------------------------------------
// WMMA bf16 GEMM with TDM-staged weights:
//   C[M x 256] = A[M x 256] @ W^T,  W row-major [256 x 256] (C[m,n]=sum_k A[m,k]W[n,k])
//
// - Whole W (256x256 bf16 = 128 KB) is DMA'd into LDS once per workgroup by the
//   Tensor Data Mover (tensor_load_to_lds), waited with s_wait_tensorcnt.
// - 8 waves / block; wave w owns a 16-row strip and all 16 N-tiles
//   (16 v8f accumulators). A is read from HBM exactly once; B fragments come
//   from LDS as ds_load_b128 pairs. 128 v_wmma per wave.
//
// Fragment layouts per CDNA5 ISA 7.12.2 (wave32):
//   A (16x32 bf16): lane L row M=L&15; VGPR r K-pair = (L>>4)*8 + (r>=4?16:0) + (r&3)*2
//   B (32x16 bf16): lane L col N=L&15; VGPR r K-pair = (L>>4)*8 + (r>=4?16:0) + (r&3)*2
//   C/D (16x16 f32): VGPR r -> M=(L>>4)*8+r, N=L&15
// ---------------------------------------------------------------------------
__global__ __launch_bounds__(256)
void gemm_bt_bf16_lds_kernel(const __bf16* __restrict__ A,
                             const __bf16* __restrict__ W,
                             float* __restrict__ C, int M) {
    extern __shared__ __bf16 Wlds[];                 // 256*256 bf16 = 128 KB

    // ---- TDM: stage all of W into LDS (issued by wave 0 only) ----
    if ((threadIdx.x >> 5) == 0) {
        unsigned long long ga = (unsigned long long)(size_t)W;     // 57-bit global
        unsigned lds_off = (unsigned)(unsigned long long)(size_t)&Wlds[0]; // LDS addr[31:0]
        // D# group 0: count=1, lds_addr, global_addr, type=2 ("image")
        v4u g0;
        g0[0] = 1u;                                   // count=1, is_restore=0, gather=0
        g0[1] = lds_off;                              // lds_addr [63:32]
        g0[2] = (unsigned)(ga & 0xFFFFFFFFu);         // global_addr [95:64]
        g0[3] = (unsigned)(ga >> 32) | (2u << 30);    // global_addr hi + type=2
        // D# group 1: data_size=2B, tensor 256x256, tile 256x256, dim0_stride=256
        v8i32 g1;
        g1[0] = (1 << 16);                            // data_size=1 (2 bytes), no flags
        g1[1] = (int)((256u & 0xFFFFu) << 16);        // tensor_dim0[15:0] in bits 63:48
        g1[2] = (int)((256u & 0xFFFFu) << 16);        // tensor_dim1[15:0] in bits 95:80
        g1[3] = (int)((256u & 0xFFFFu) << 16);        // tile_dim0 in bits 127:112
        g1[4] = 256;                                  // tile_dim1 (bits 143:128), tile_dim2=0
        g1[5] = 256;                                  // tensor_dim0_stride low 32
        g1[6] = 0;
        g1[7] = 0;
        v4i32 gz4 = {0, 0, 0, 0};                     // groups 2/3 unused (2-D tensor)
        v8i32 gz8 = {0, 0, 0, 0, 0, 0, 0, 0};
        __builtin_amdgcn_tensor_load_to_lds(g0, g1, gz4, gz4, gz8, 0);
        __builtin_amdgcn_s_wait_tensorcnt(0);
    }
    __syncthreads();

    const int waveInBlk = threadIdx.x >> 5;
    const int lane = threadIdx.x & 31;
    const int tileM = blockIdx.x * 8 + waveInBlk;     // 16-row strip per wave
    const int half = lane >> 4;
    const int l16  = lane & 15;

    const unsigned* Arow  = (const unsigned*)(A + (size_t)(tileM * 16 + l16) * DIN);
    const unsigned* WldsU = (const unsigned*)&Wlds[0];

    union Frag { v4u q[2]; v16bf v; };

    v8f acc[16];
#pragma unroll
    for (int nt = 0; nt < 16; ++nt) acc[nt] = (v8f){};

    for (int k0 = 0; k0 < DIN; k0 += 32) {
        Frag af;
        const unsigned* ap = Arow + (k0 >> 1) + half * 4;
        af.q[0] = *(const v4u*)(ap);
        af.q[1] = *(const v4u*)(ap + 8);
#pragma unroll
        for (int nt = 0; nt < 16; ++nt) {
            const unsigned* wrow = WldsU + (((nt * 16 + l16) << 7) + (k0 >> 1) + (half << 2));
            Frag bf;
            bf.q[0] = *(const v4u*)(wrow);
            bf.q[1] = *(const v4u*)(wrow + 8);
            acc[nt] = __builtin_amdgcn_wmma_f32_16x16x32_bf16(
                          false, af.v, false, bf.v, (short)0, acc[nt], false, false);
        }
    }

#pragma unroll
    for (int nt = 0; nt < 16; ++nt) {
#pragma unroll
        for (int r = 0; r < 8; ++r) {
            int m = tileM * 16 + half * 8 + r;
            int n = nt * 16 + l16;
            C[(size_t)m * DIN + n] = acc[nt][r];
        }
    }
}

// ---------------------------------------------------------------------------
// Additive-attention scores + softmax over the Q axis (size 8).
// score[bh,q,k] = sum_d tanh(q[bh,q,d] + k[bh,k,d]) * wv[d]
// softmax is over q per (bh,k): fully streaming in k.
// ---------------------------------------------------------------------------
__global__ void attn_score_softmax_kernel(const float* __restrict__ qproj,
                                          const float* __restrict__ kproj,
                                          const float* __restrict__ wv,
                                          float* __restrict__ attn) {
    int bh = blockIdx.x;                 // 0..255
    int b = bh >> 3, h = bh & 7;
    __shared__ float qs[QLEN][DH];
    __shared__ float ws[DH];
    int t = threadIdx.x;                 // 256 threads
    {
        int q = t >> 5, d = t & 31;
        qs[q][d] = qproj[((size_t)b * QLEN + q) * DIN + h * DH + d];
        if (t < DH) ws[t] = wv[t];
    }
    __syncthreads();

    for (int k = t; k < KLEN; k += 256) {
        const float* kv = &kproj[((size_t)b * KLEN + k) * DIN + h * DH];
        float kd[DH];
#pragma unroll
        for (int d = 0; d < DH; ++d) kd[d] = kv[d];

        float sc[QLEN];
#pragma unroll
        for (int q = 0; q < QLEN; ++q) {
            float s = 0.f;
#pragma unroll
            for (int d = 0; d < DH; ++d)
                s += fast_tanh(qs[q][d] + kd[d]) * ws[d];
            sc[q] = s;
        }
        float mx = sc[0];
#pragma unroll
        for (int q = 1; q < QLEN; ++q) mx = fmaxf(mx, sc[q]);
        float sum = 0.f;
#pragma unroll
        for (int q = 0; q < QLEN; ++q) { sc[q] = fast_exp(sc[q] - mx); sum += sc[q]; }
        float inv = __builtin_amdgcn_rcpf(sum);
#pragma unroll
        for (int q = 0; q < QLEN; ++q)
            attn[((size_t)bh * QLEN + q) * KLEN + k] = sc[q] * inv;
    }
}

// ---------------------------------------------------------------------------
// out[b,q,h*32+d] = sum_k attn[bh,q,k] * v[b,k,h*32+d]     (merged-head layout)
// ---------------------------------------------------------------------------
__global__ void attn_v_kernel(const float* __restrict__ attn,
                              const float* __restrict__ vproj,
                              float* __restrict__ out) {
    int bh = blockIdx.x;
    int b = bh >> 3, h = bh & 7;
    int t = threadIdx.x;
    int q = t >> 5, d = t & 31;
    const float* arow = &attn[((size_t)bh * QLEN + q) * KLEN];
    const float* vcol = &vproj[(size_t)b * KLEN * DIN + h * DH + d];
    float a0 = 0.f, a1 = 0.f, a2 = 0.f, a3 = 0.f;
    for (int k = 0; k < KLEN; k += 4) {
        a0 += arow[k + 0] * vcol[(size_t)(k + 0) * DIN];
        a1 += arow[k + 1] * vcol[(size_t)(k + 1) * DIN];
        a2 += arow[k + 2] * vcol[(size_t)(k + 2) * DIN];
        a3 += arow[k + 3] * vcol[(size_t)(k + 3) * DIN];
    }
    out[((size_t)b * QLEN + q) * DIN + h * DH + d] = (a0 + a1) + (a2 + a3);
}

// ---------------------------------------------------------------------------
// final[b,n] = sum_{j<2048} tmp[b,j] * fcW[n,j] + fcb[n]
// ---------------------------------------------------------------------------
__global__ void final_fc_kernel(const float* __restrict__ tmp,   // (32, 2048)
                                const float* __restrict__ fcW,   // (256, 2048)
                                const float* __restrict__ fcb,   // (256)
                                float* __restrict__ out) {       // (32, 256)
    int wave = (blockIdx.x * blockDim.x + threadIdx.x) >> 5;
    int lane = threadIdx.x & 31;
    int b = wave >> 8;
    int n = wave & 255;
    const float* x = &tmp[(size_t)b * 2048];
    const float* w = &fcW[(size_t)n * 2048];
    float acc = 0.f;
    for (int j = lane; j < 2048; j += 32) acc += x[j] * w[j];
#pragma unroll
    for (int off = 16; off > 0; off >>= 1) acc += __shfl_down(acc, off, 32);
    if (lane == 0) out[(size_t)b * 256 + n] = acc + fcb[n];
}

// ---------------------------------------------------------------------------
// host launcher
// ---------------------------------------------------------------------------
static inline size_t alignUp(size_t x, size_t a) { return (x + a - 1) & ~(a - 1); }

extern "C" void kernel_launch(void* const* d_in, const int* in_sizes, int n_in,
                              void* d_out, int out_size, void* d_ws, size_t ws_size,
                              hipStream_t stream) {
    const float* queries = (const float*)d_in[0];   // (32, 8, 256)
    const float* keys    = (const float*)d_in[1];   // (32, 2048, 256)
    const float* values  = (const float*)d_in[2];   // (32, 2048, 256)
    const float* Wq      = (const float*)d_in[3];   // (256, 256)
    const float* Wk      = (const float*)d_in[4];
    const float* Wv      = (const float*)d_in[5];
    const float* Wo      = (const float*)d_in[6];
    const float* wvs     = (const float*)d_in[7];   // (32,)
    const float* fcW     = (const float*)d_in[8];   // (256, 2048)
    const float* fcb     = (const float*)d_in[9];   // (256,)
    float* out           = (float*)d_out;           // (32, 256)

    const size_t nQ  = (size_t)BATCH * QLEN * DIN;   // 65536
    const size_t nKV = (size_t)BATCH * KLEN * DIN;   // 16777216
    const size_t nW  = (size_t)DIN * DIN;            // 65536

    char* p = (char*)d_ws;
    size_t off = 0;
    auto take = [&](size_t bytes) { void* r = p + off; off = alignUp(off + bytes, 256); return r; };
    __bf16* q_bf   = (__bf16*)take(nQ  * 2);
    __bf16* k_bf   = (__bf16*)take(nKV * 2);
    __bf16* v_bf   = (__bf16*)take(nKV * 2);
    __bf16* wq_bf  = (__bf16*)take(nW * 2);
    __bf16* wk_bf  = (__bf16*)take(nW * 2);
    __bf16* wv_bf  = (__bf16*)take(nW * 2);
    __bf16* wo_bf  = (__bf16*)take(nW * 2);
    float*  qproj  = (float*)take(nQ  * 4);
    float*  kproj  = (float*)take(nKV * 4);
    float*  vproj  = (float*)take(nKV * 4);
    float*  attn   = (float*)take((size_t)BH * QLEN * KLEN * 4);   // 16 MB
    float*  aout   = (float*)take(nQ * 4);
    __bf16* aout_bf= (__bf16*)take(nQ * 2);
    float*  tmp    = (float*)take(nQ * 4);
    (void)ws_size;

    // 1) convert inputs/weights to bf16
    auto cvt = [&](const float* src, __bf16* dst, size_t n) {
        int blocks = (int)((n + 255) / 256);
        f32_to_bf16_kernel<<<blocks, 256, 0, stream>>>(src, dst, (int)n);
    };
    cvt(queries, q_bf, nQ);
    cvt(keys,    k_bf, nKV);
    cvt(values,  v_bf, nKV);
    cvt(Wq, wq_bf, nW);
    cvt(Wk, wk_bf, nW);
    cvt(Wv, wv_bf, nW);
    cvt(Wo, wo_bf, nW);

    // 2) projections: C = A @ W^T  (WMMA bf16, TDM-staged W in LDS)
    auto gemm = [&](const __bf16* A, const __bf16* W, float* C, int M) {
        int blocks = M / 128;                         // 8 waves x 16 rows per block
        gemm_bt_bf16_lds_kernel<<<blocks, 256, 128 * 1024, stream>>>(A, W, C, M);
    };
    gemm(q_bf, wq_bf, qproj, BATCH * QLEN);           // 256 rows
    gemm(k_bf, wk_bf, kproj, BATCH * KLEN);           // 65536 rows
    gemm(v_bf, wv_bf, vproj, BATCH * KLEN);

    // 3) additive attention scores + softmax over q
    attn_score_softmax_kernel<<<BH, 256, 0, stream>>>(qproj, kproj, wvs, attn);

    // 4) attn @ v  -> merged-head output (B, Q, 256)
    attn_v_kernel<<<BH, 256, 0, stream>>>(attn, vproj, aout);

    // 5) out @ Wo^T  (WMMA again)
    cvt(aout, aout_bf, nQ);
    gemm(aout_bf, wo_bf, tmp, BATCH * QLEN);

    // 6) final FC: (32, 2048) @ fcW^T + fcb -> (32, 256)
    {
        int waves = BATCH * 256;                      // 8192 waves
        int blocks = waves / 8;
        final_fc_kernel<<<blocks, 256, 0, stream>>>(tmp, fcW, fcb, out);
    }
}
